// RoPESelfAttention_58076547776991
// MI455X (gfx1250) — matrix operand order repaired and checked by reading, not compiled, not run on hardware
//
#include <hip/hip_runtime.h>
#include <hip/hip_bf16.h>

// ---------------------------------------------------------------------------
// CDNA5 (gfx1250) wave32 WMMA implementation of RoPE self-attention.
// All matmuls use v_wmma_f32_16x16x32_bf16. Flash-attention (online softmax)
// avoids materializing the [B,H,S,S] score tensor (1 GB of HBM traffic).
// ---------------------------------------------------------------------------

typedef __attribute__((ext_vector_type(16))) __bf16 v16bf;
typedef __attribute__((ext_vector_type(8)))  float  v8f;

union FragBF {
  v16bf v;
  uint4 q[2];
};

static constexpr int Bc  = 4;
static constexpr int Sc  = 2048;
static constexpr int Dc  = 1024;
static constexpr int Hc  = 16;
static constexpr int HDc = 64;

__device__ __forceinline__ unsigned short f2bf(float f) {
  union { float f; unsigned u; } x;
  x.f = f;
  unsigned u = x.u;
  unsigned r = (u + 0x7FFFu + ((u >> 16) & 1u)) >> 16;  // round-to-nearest-even
  return (unsigned short)r;
}

// ---------------------------------------------------------------------------
// fp32 -> bf16 conversion (grid-stride)
// ---------------------------------------------------------------------------
__global__ void cvt_f32_bf16(const float* __restrict__ s,
                             unsigned short* __restrict__ d, int n) {
  int i = blockIdx.x * blockDim.x + threadIdx.x;
  int stride = gridDim.x * blockDim.x;
  for (; i < n; i += stride) d[i] = f2bf(s[i]);
}

// ---------------------------------------------------------------------------
// GEMM: C[M,N] = A[M,K](bf16) @ W[N,K]^T(bf16) + bias, M=8192, N=K=1024.
// Wave computes a 16x64 tile (4 accumulator tiles along N). Since HD==64 the
// wave's 64 output columns are exactly one attention head.
//   MODE 0: plain + bias, fp32 store to out [M,N]           (O projection)
//   MODE 1: bias + RoPE,  bf16 store to [B,H,S,HD]          (Q and K)
//   MODE 2: bias,         bf16 transposed store to [B,H,HD,S] (V)
// ---------------------------------------------------------------------------
template <int MODE>
__global__ __launch_bounds__(128) void gemm_bf16_wmma(
    const unsigned short* __restrict__ A,   // [M,K] bf16
    const unsigned short* __restrict__ W,   // [N,K] bf16
    const float* __restrict__ bias,         // [N]
    const float* __restrict__ cosp,         // [S,HD] (MODE 1)
    const float* __restrict__ sinp,         // [S,HD] (MODE 1)
    float* __restrict__ outF,               // MODE 0
    unsigned short* __restrict__ outB)      // MODE 1/2
{
  constexpr int K = Dc, N = Dc;
  const int lane = threadIdx.x & 31;
  const int wv   = threadIdx.x >> 5;
  const int hi   = lane >> 4;   // 0: lanes 0-15, 1: lanes 16-31
  const int l15  = lane & 15;
  const int m0   = blockIdx.y * 64 + wv * 16;
  const int n0   = blockIdx.x * 64;

  const v8f vzero = {0.f,0.f,0.f,0.f,0.f,0.f,0.f,0.f};
  v8f acc[4];
#pragma unroll
  for (int t = 0; t < 4; ++t) acc[t] = vzero;

  // A fragment (16x32, contiguous k order used for A and B here): lane = M row; lanes 0-15 hold K 0..15,
  // lanes 16-31 hold K 16..31.
  const unsigned short* arow = A + (size_t)(m0 + l15) * K;

  for (int k = 0; k < K; k += 32) {
    __builtin_prefetch(arow + k + 128, 0, 0);  // global_prefetch_b8
    FragBF fa;
    fa.q[0] = *(const uint4*)(arow + k + hi * 16);
    fa.q[1] = *(const uint4*)(arow + k + hi * 16 + 8);
#pragma unroll
    for (int t = 0; t < 4; ++t) {
      // B fragment (32x16, same contiguous k order as A): lane = N col (W row), lanes 0-15 hold
      // K 0..15, lanes 16-31 hold K 16..31 -> one contiguous 32B read of W.
      const unsigned short* wrow =
          W + (size_t)(n0 + 16 * t + l15) * K + k + hi * 16;
      FragBF fb;
      fb.q[0] = *(const uint4*)(wrow);
      fb.q[1] = *(const uint4*)(wrow + 8);
      acc[t] = __builtin_amdgcn_wmma_f32_16x16x32_bf16(
          false, fa.v, false, fb.v, (short)0, acc[t], false, false);
    }
  }

  // C layout: VGPR r, lanes 0-15 -> row r, lanes 16-31 -> row r+8; col = lane&15.
  if constexpr (MODE == 0) {
#pragma unroll
    for (int t = 0; t < 4; ++t) {
      const int col = n0 + 16 * t + l15;
      const float bv = bias[col];
#pragma unroll
      for (int r = 0; r < 8; ++r) {
        const int row = m0 + r + hi * 8;
        outF[(size_t)row * N + col] = acc[t][r] + bv;
      }
    }
  } else if constexpr (MODE == 1) {
    const int bb = m0 / Sc;
    const int sbase = m0 % Sc;
    const int h = n0 >> 6;  // HD == 64
#pragma unroll
    for (int t = 0; t < 2; ++t) {
      const int i1 = 16 * t + l15;     // first-half rotary index
      const int i2 = i1 + 32;          // paired second-half index
      const float b1 = bias[n0 + i1];
      const float b2 = bias[n0 + i2];
#pragma unroll
      for (int r = 0; r < 8; ++r) {
        const int s = sbase + r + hi * 8;
        const float c1 = cosp[s * HDc + i1], s1 = sinp[s * HDc + i1];
        const float c2 = cosp[s * HDc + i2], s2 = sinp[s * HDc + i2];
        const float x1 = acc[t][r] + b1;
        const float x2 = acc[t + 2][r] + b2;
        const float o1 = x1 * c1 - x2 * s1;   // x*cos + rotate_half(x)*sin
        const float o2 = x2 * c2 + x1 * s2;
        const size_t base = (((size_t)bb * Hc + h) * Sc + s) * HDc;
        outB[base + i1] = f2bf(o1);
        outB[base + i2] = f2bf(o2);
      }
    }
  } else {  // MODE 2: V transposed to [B,H,HD,S]
    const int bb = m0 / Sc;
    const int sbase = m0 % Sc;
    const int h = n0 >> 6;
#pragma unroll
    for (int t = 0; t < 4; ++t) {
      const int i = 16 * t + l15;
      const float bv = bias[n0 + i];
#pragma unroll
      for (int r = 0; r < 8; ++r) {
        const int s = sbase + r + hi * 8;
        outB[(((size_t)bb * Hc + h) * HDc + i) * Sc + s] = f2bf(acc[t][r] + bv);
      }
    }
  }
}

// ---------------------------------------------------------------------------
// Flash attention: one wave per 16-query tile, loop over 32 keys per step.
// Q,K bf16 [B,H,S,HD]; V bf16 [B,H,HD,S]; ctx bf16 [B,S,D].
// ---------------------------------------------------------------------------
__global__ __launch_bounds__(128) void attn_wmma(
    const unsigned short* __restrict__ Qm,
    const unsigned short* __restrict__ Km,
    const unsigned short* __restrict__ Vt,
    const float* __restrict__ mask,       // [B,S]
    unsigned short* __restrict__ ctx)     // [B,S,D]
{
  __shared__ __align__(16) unsigned short plds[4][16 * 32];  // per-wave P tile

  const int lane = threadIdx.x & 31;
  const int wv   = threadIdx.x >> 5;
  const int hi   = lane >> 4;
  const int l15  = lane & 15;
  const int bh   = blockIdx.y;
  const int b    = bh >> 4;   // H == 16
  const int h    = bh & 15;
  const int q0   = blockIdx.x * 64 + wv * 16;

  const unsigned short* qb = Qm + (size_t)bh * Sc * HDc;
  const unsigned short* kb = Km + (size_t)bh * Sc * HDc;
  const unsigned short* vb = Vt + (size_t)bh * HDc * Sc;

  // Q fragments for K-dim 0..31 and 32..63 (A-layout)
  FragBF qa[2];
  const unsigned short* qrow = qb + (size_t)(q0 + l15) * HDc;
#pragma unroll
  for (int kk = 0; kk < 2; ++kk) {
    qa[kk].q[0] = *(const uint4*)(qrow + kk * 32 + hi * 16);
    qa[kk].q[1] = *(const uint4*)(qrow + kk * 32 + hi * 16 + 8);
  }

  const v8f vzero = {0.f,0.f,0.f,0.f,0.f,0.f,0.f,0.f};
  v8f o[4];
#pragma unroll
  for (int t = 0; t < 4; ++t) o[t] = vzero;
  float mrow[8], lrow[8];
#pragma unroll
  for (int r = 0; r < 8; ++r) { mrow[r] = -3.0e38f; lrow[r] = 0.f; }

  const float scale = 0.125f;  // 1/sqrt(64)

  for (int j0 = 0; j0 < Sc; j0 += 32) {
    // ---- scores: two 16x16 tiles (keys j0..+15 and j0+16..+31) ----
    v8f s0 = vzero, s1 = vzero;
#pragma unroll
    for (int kk = 0; kk < 2; ++kk) {
      const int ko = kk * 32;
      const unsigned short* kr0 = kb + (size_t)(j0 + l15) * HDc + ko + hi * 16;
      FragBF fk0;
      fk0.q[0] = *(const uint4*)kr0;
      fk0.q[1] = *(const uint4*)(kr0 + 8);
      s0 = __builtin_amdgcn_wmma_f32_16x16x32_bf16(
          false, qa[kk].v, false, fk0.v, (short)0, s0, false, false);
      const unsigned short* kr1 =
          kb + (size_t)(j0 + 16 + l15) * HDc + ko + hi * 16;
      FragBF fk1;
      fk1.q[0] = *(const uint4*)kr1;
      fk1.q[1] = *(const uint4*)(kr1 + 8);
      s1 = __builtin_amdgcn_wmma_f32_16x16x32_bf16(
          false, qa[kk].v, false, fk1.v, (short)0, s1, false, false);
    }

    const float mb0 = (1.f - mask[b * Sc + j0 + l15]) * -1e9f;
    const float mb1 = (1.f - mask[b * Sc + j0 + 16 + l15]) * -1e9f;

    // ---- online softmax (row r of the C tile lives across 16 lanes) ----
#pragma unroll
    for (int r = 0; r < 8; ++r) {
      float v0 = s0[r] * scale + mb0;
      float v1 = s1[r] * scale + mb1;
      float mx = fmaxf(v0, v1);
#pragma unroll
      for (int off = 1; off < 16; off <<= 1)
        mx = fmaxf(mx, __shfl_xor(mx, off, 16));
      const float mnew  = fmaxf(mrow[r], mx);
      const float alpha = __expf(mrow[r] - mnew);
      const float p0 = __expf(v0 - mnew);
      const float p1 = __expf(v1 - mnew);
      float rs = p0 + p1;
#pragma unroll
      for (int off = 1; off < 16; off <<= 1) rs += __shfl_xor(rs, off, 16);
      lrow[r] = lrow[r] * alpha + rs;
      mrow[r] = mnew;
#pragma unroll
      for (int t = 0; t < 4; ++t) o[t][r] *= alpha;
      const int row = r + hi * 8;
      plds[wv][row * 32 + l15]      = f2bf(p0);
      plds[wv][row * 32 + 16 + l15] = f2bf(p1);
    }
    __syncthreads();  // C-layout -> A-layout for P goes through LDS

    FragBF pf;
    pf.q[0] = *(const uint4*)&plds[wv][l15 * 32 + hi * 16];
    pf.q[1] = *(const uint4*)&plds[wv][l15 * 32 + hi * 16 + 8];

    // ---- O += P(16x32) @ V(32x64); V is [HD,S] so B-frags are contiguous ----
#pragma unroll
    for (int t = 0; t < 4; ++t) {
      const unsigned short* vr =
          vb + (size_t)(16 * t + l15) * Sc + j0 + hi * 16;
      FragBF fv;
      fv.q[0] = *(const uint4*)vr;
      fv.q[1] = *(const uint4*)(vr + 8);
      o[t] = __builtin_amdgcn_wmma_f32_16x16x32_bf16(
          false, pf.v, false, fv.v, (short)0, o[t], false, false);
    }
    __syncthreads();
  }

  // ---- epilogue: normalize, store bf16 context [B,S,D] ----
#pragma unroll
  for (int t = 0; t < 4; ++t) {
    const int col = h * HDc + 16 * t + l15;
#pragma unroll
    for (int r = 0; r < 8; ++r) {
      const int s = q0 + r + hi * 8;
      const float val = o[t][r] / lrow[r];
      ctx[((size_t)(b * Sc + s)) * Dc + col] = f2bf(val);
    }
  }
}

// ---------------------------------------------------------------------------
// Host launcher
// ---------------------------------------------------------------------------
extern "C" void kernel_launch(void* const* d_in, const int* in_sizes, int n_in,
                              void* d_out, int out_size, void* d_ws,
                              size_t ws_size, hipStream_t stream) {
  const float* x    = (const float*)d_in[0];
  const float* cosp = (const float*)d_in[1];
  const float* sinp = (const float*)d_in[2];
  const float* mask = (const float*)d_in[3];
  const float* q_w  = (const float*)d_in[4];
  const float* q_b  = (const float*)d_in[5];
  const float* k_w  = (const float*)d_in[6];
  const float* k_b  = (const float*)d_in[7];
  const float* v_w  = (const float*)d_in[8];
  const float* v_b  = (const float*)d_in[9];
  const float* o_w  = (const float*)d_in[10];
  const float* o_b  = (const float*)d_in[11];

  const size_t M  = (size_t)Bc * Sc;   // 8192
  const size_t MD = M * Dc;            // 8,388,608
  const size_t DD = (size_t)Dc * Dc;   // 1,048,576

  unsigned short* xw  = (unsigned short*)d_ws;      // x bf16       [M,D]
  unsigned short* qw  = xw + MD;                    // q_w bf16     [D,D]
  unsigned short* kw  = qw + DD;
  unsigned short* vw  = kw + DD;
  unsigned short* ow  = vw + DD;
  unsigned short* qB  = ow + DD;                    // Q+rope bf16  [B,H,S,HD]
  unsigned short* kB  = qB + MD;                    // K+rope bf16  [B,H,S,HD]
  unsigned short* vT  = kB + MD;                    // V^T bf16     [B,H,HD,S]
  unsigned short* ctx = vT + MD;                    // context bf16 [B,S,D]

  // fp32 -> bf16 conversions
  cvt_f32_bf16<<<4096, 256, 0, stream>>>(x, xw, (int)MD);
  cvt_f32_bf16<<<1024, 256, 0, stream>>>(q_w, qw, (int)DD);
  cvt_f32_bf16<<<1024, 256, 0, stream>>>(k_w, kw, (int)DD);
  cvt_f32_bf16<<<1024, 256, 0, stream>>>(v_w, vw, (int)DD);
  cvt_f32_bf16<<<1024, 256, 0, stream>>>(o_w, ow, (int)DD);

  // QKV projections (RoPE fused for Q/K, transpose fused for V)
  dim3 gg(Dc / 64, (unsigned)(M / 64));  // (16, 128), 128 threads = 4 waves
  gemm_bf16_wmma<1><<<gg, 128, 0, stream>>>(xw, qw, q_b, cosp, sinp, nullptr, qB);
  gemm_bf16_wmma<1><<<gg, 128, 0, stream>>>(xw, kw, k_b, cosp, sinp, nullptr, kB);
  gemm_bf16_wmma<2><<<gg, 128, 0, stream>>>(xw, vw, v_b, nullptr, nullptr, nullptr, vT);

  // flash attention
  dim3 ga(Sc / 64, Bc * Hc);  // (32, 64)
  attn_wmma<<<ga, 128, 0, stream>>>(qB, kB, vT, mask, ctx);

  // output projection -> fp32 d_out
  gemm_bf16_wmma<0><<<gg, 128, 0, stream>>>(ctx, ow, o_b, nullptr, nullptr,
                                            (float*)d_out, nullptr);
}
